// SelfAttention_47983374631448
// MI455X (gfx1250) — compile-verified, matrix operand order checked by reading, not run on hardware
//
#include <hip/hip_runtime.h>
#include <hip/hip_bf16.h>

// ---------------------------------------------------------------------------
// Self-attention forward (batch 0 only) for MI455X / gfx1250, wave32 WMMA.
//   x[0] (2048x1024 f32) -> Q,K,V proj (f16 WMMA, f32 accum) -> flash attn
//   per head (online softmax) -> output proj -> d_out f32 [2048,1024]
// Register double-buffered fragment loads to overlap VMEM with v_wmma.
// ---------------------------------------------------------------------------

#define S_LEN 2048
#define HIDN  1024
#define NHEAD 16
#define HDIM  64

typedef _Float16 v8h  __attribute__((ext_vector_type(8)));
typedef _Float16 v16h __attribute__((ext_vector_type(16)));
typedef float    v8f  __attribute__((ext_vector_type(8)));

__device__ __forceinline__ v8f zero8() {
    v8f z;
#pragma unroll
    for (int i = 0; i < 8; ++i) z[i] = 0.0f;
    return z;
}

// 16x32 f16 fragment load (A-layout; B uses the mirrored layout with the same
// lane mapping: lane L -> row/col (L&15), K-halves selected by lane bit 4).
// Two contiguous 16-byte loads per lane.
__device__ __forceinline__ v16h load_frag(const _Float16* __restrict__ base,
                                          int row, int k, int ld) {
    const int lane = threadIdx.x & 31;
    const int r    = lane & 15;
    const int hi   = lane >> 4;
    const _Float16* p = base + (size_t)(row + r) * (size_t)ld + k + hi * 8;
    v8h lo = *(const v8h*)p;
    v8h hh = *(const v8h*)(p + 16);
    return __builtin_shufflevector(lo, hh, 0, 1, 2, 3, 4, 5, 6, 7,
                                   8, 9, 10, 11, 12, 13, 14, 15);
}

__device__ __forceinline__ v8f wmma_f16(v16h a, v16h b, v8f c) {
    // D = A x B + C, f16 inputs, f32 accumulate.
    return __builtin_amdgcn_wmma_f32_16x16x32_f16(
        /*neg_a=*/false, a, /*neg_b=*/false, b,
        /*c_mod=*/(short)0, c, /*reuse_a=*/false, /*reuse_b=*/false);
}

// ---------------------------------------------------------------------------
// f32 -> f16 conversion
// ---------------------------------------------------------------------------
__global__ void cvt_f32_to_f16(const float* __restrict__ src,
                               _Float16* __restrict__ dst, int n) {
    int i = blockIdx.x * blockDim.x + threadIdx.x;
    if (i < n) dst[i] = (_Float16)src[i];
}

// ---------------------------------------------------------------------------
// C[M,N] = A[M,K] * W[N,K]^T + bias, WMMA tiled. 256 threads = 8 waves; each
// wave owns a 16x64 tile (A fragment reused across 4 B fragments / WMMAs).
// Fragments are double-buffered in registers: loads for k+32 are issued
// before the WMMAs for k so VMEM latency overlaps matrix ops.
// mode 0: f32 row-major out (final projection)
// mode 1: f16 out, per-head [h][row][d]     (Q with scale=0.125, K)
// mode 2: f16 out, per-head transposed [h][d][row]  (V)
// ---------------------------------------------------------------------------
__global__ void gemm_wmma(const _Float16* __restrict__ A,
                          const _Float16* __restrict__ W,
                          const float* __restrict__ bias,
                          void* __restrict__ outp,
                          int M, int N, int K, int mode, float scale) {
    const int wid  = threadIdx.x >> 5;
    const int lane = threadIdx.x & 31;
    const int r    = lane & 15;
    const int hi   = lane >> 4;
    const int mbase = blockIdx.y * 64 + (wid & 3) * 16;
    const int nbase = blockIdx.x * 128 + (wid >> 2) * 64;

    v8f acc[4] = {zero8(), zero8(), zero8(), zero8()};

    // prologue: fragments for k = 0
    v16h a_cur = load_frag(A, mbase, 0, K);
    v16h b_cur[4];
#pragma unroll
    for (int t = 0; t < 4; ++t) b_cur[t] = load_frag(W, nbase + t * 16, 0, K);

    for (int k = 0; k < K; k += 32) {
        const int kn = (k + 32 < K) ? (k + 32) : 0;  // last iter reloads k=0 (discarded)
        // issue next-step loads first so they overlap this step's WMMAs
        v16h a_nxt = load_frag(A, mbase, kn, K);
        v16h b_nxt[4];
#pragma unroll
        for (int t = 0; t < 4; ++t) b_nxt[t] = load_frag(W, nbase + t * 16, kn, K);
        __builtin_prefetch(A + (size_t)(mbase + r) * (size_t)K + kn + 32, 0, 3);

#pragma unroll
        for (int t = 0; t < 4; ++t) acc[t] = wmma_f16(a_cur, b_cur[t], acc[t]);

        a_cur = a_nxt;
#pragma unroll
        for (int t = 0; t < 4; ++t) b_cur[t] = b_nxt[t];
    }

#pragma unroll
    for (int t = 0; t < 4; ++t) {
        const int col = nbase + t * 16 + r;
        const float bv = bias[col];
#pragma unroll
        for (int rr = 0; rr < 8; ++rr) {
            const int row = mbase + rr + hi * 8;
            const float val = (acc[t][rr] + bv) * scale;
            if (mode == 0) {
                ((float*)outp)[(size_t)row * N + col] = val;
            } else {
                const int h = col >> 6;
                const int d = col & 63;
                _Float16* o = (_Float16*)outp;
                if (mode == 1)
                    o[((size_t)h * M + row) * HDIM + d] = (_Float16)val;
                else
                    o[((size_t)h * HDIM + d) * M + row] = (_Float16)val;
            }
        }
    }
}

// ---------------------------------------------------------------------------
// Fused per-head flash attention. Grid (S/64, NH), block 128 = 4 waves; each
// wave streams 16 query rows over all keys in 32-key tiles:
//   S = Q*K^T (4 WMMAs) -> issue V loads + next-K loads -> online softmax
//   (VALU work hides the in-flight loads) -> P via LDS relayout ->
//   ctx += P*V (4 WMMAs against transposed V)
// Q is pre-scaled by 1/sqrt(HD). Output ctx f16 [S][HID] row-major.
// ---------------------------------------------------------------------------
__global__ void attn_flash(const _Float16* __restrict__ Qg,
                           const _Float16* __restrict__ Kg,
                           const _Float16* __restrict__ Vtg,
                           _Float16* __restrict__ ctx) {
    const int h    = blockIdx.y;
    const int wid  = threadIdx.x >> 5;
    const int lane = threadIdx.x & 31;
    const int r    = lane & 15;
    const int hi   = lane >> 4;
    const int qrow = blockIdx.x * 64 + wid * 16;

    const _Float16* Qh = Qg  + (size_t)h * S_LEN * HDIM;
    const _Float16* Kh = Kg  + (size_t)h * S_LEN * HDIM;
    const _Float16* Vh = Vtg + (size_t)h * HDIM * S_LEN;   // [d][s]

    __shared__ __align__(16) _Float16 plds[4][16 * 32];    // per-wave P tile
    _Float16* myp = &plds[wid][0];

    const v16h aq0 = load_frag(Qh, qrow, 0, HDIM);
    const v16h aq1 = load_frag(Qh, qrow, 32, HDIM);

    v8f acc[4] = {zero8(), zero8(), zero8(), zero8()};
    float mi[8], li[8];
#pragma unroll
    for (int rr = 0; rr < 8; ++rr) { mi[rr] = -1.0e30f; li[rr] = 0.0f; }

    // prologue: K fragments for kb = 0
    v16h kf[4];
    kf[0] = load_frag(Kh, 0, 0, HDIM);
    kf[1] = load_frag(Kh, 0, 32, HDIM);
    kf[2] = load_frag(Kh, 16, 0, HDIM);
    kf[3] = load_frag(Kh, 16, 32, HDIM);

    for (int kb = 0; kb < S_LEN; kb += 32) {
        // ---- scores: 16q x 32k (two 16x16 tiles, K-dim = head dim 64) ----
        v8f s0 = wmma_f16(aq0, kf[0], zero8());
        s0 = wmma_f16(aq1, kf[1], s0);
        v8f s1 = wmma_f16(aq0, kf[2], zero8());
        s1 = wmma_f16(aq1, kf[3], s1);

        // ---- issue V loads (this tile) and K loads (next tile) now; the
        //      softmax VALU section below hides their latency ----
        v16h vf[4];
#pragma unroll
        for (int t4 = 0; t4 < 4; ++t4) vf[t4] = load_frag(Vh, t4 * 16, kb, S_LEN);
        const int kn = (kb + 32 < S_LEN) ? (kb + 32) : 0;
        kf[0] = load_frag(Kh, kn, 0, HDIM);
        kf[1] = load_frag(Kh, kn, 32, HDIM);
        kf[2] = load_frag(Kh, kn + 16, 0, HDIM);
        kf[3] = load_frag(Kh, kn + 16, 32, HDIM);

        // ---- online softmax (rows live across 16-lane halves) ----
#pragma unroll
        for (int rr = 0; rr < 8; ++rr) {
            const float a = s0[rr];
            const float b = s1[rr];
            float t = fmaxf(a, b);
            t = fmaxf(t, __shfl_xor(t, 1));
            t = fmaxf(t, __shfl_xor(t, 2));
            t = fmaxf(t, __shfl_xor(t, 4));
            t = fmaxf(t, __shfl_xor(t, 8));
            const float mn    = fmaxf(mi[rr], t);
            const float alpha = __expf(mi[rr] - mn);
            mi[rr] = mn;
            const float p0 = __expf(a - mn);
            const float p1 = __expf(b - mn);
            float ssum = p0 + p1;
            ssum += __shfl_xor(ssum, 1);
            ssum += __shfl_xor(ssum, 2);
            ssum += __shfl_xor(ssum, 4);
            ssum += __shfl_xor(ssum, 8);
            li[rr] = li[rr] * alpha + ssum;
#pragma unroll
            for (int t4 = 0; t4 < 4; ++t4) acc[t4][rr] *= alpha;
            // C-layout -> A-layout relayout through LDS
            const int lrow = (rr + hi * 8) * 32 + r;
            myp[lrow]      = (_Float16)p0;
            myp[lrow + 16] = (_Float16)p1;
        }

        // ---- ctx += P[16x32] * V[32x16] per 16-wide d tile ----
        const v16h ap = load_frag(myp, 0, 0, 32);
#pragma unroll
        for (int t4 = 0; t4 < 4; ++t4) acc[t4] = wmma_f16(ap, vf[t4], acc[t4]);
    }

    // ---- normalize and write ctx [S][HID] (head-interleaved) ----
#pragma unroll
    for (int rr = 0; rr < 8; ++rr) {
        const float inv = 1.0f / li[rr];
        const int row = qrow + rr + hi * 8;
        _Float16* o = ctx + (size_t)row * HIDN + h * HDIM + r;
        o[0]  = (_Float16)(acc[0][rr] * inv);
        o[16] = (_Float16)(acc[1][rr] * inv);
        o[32] = (_Float16)(acc[2][rr] * inv);
        o[48] = (_Float16)(acc[3][rr] * inv);
    }
}

// ---------------------------------------------------------------------------
extern "C" void kernel_launch(void* const* d_in, const int* in_sizes, int n_in,
                              void* d_out, int out_size, void* d_ws, size_t ws_size,
                              hipStream_t stream) {
    (void)in_sizes; (void)n_in; (void)out_size; (void)ws_size;

    const float* x  = (const float*)d_in[0];  // [2,2048,1024]; only batch 0 used
    const float* Wq = (const float*)d_in[1];
    const float* bq = (const float*)d_in[2];
    const float* Wk = (const float*)d_in[3];
    const float* bk = (const float*)d_in[4];
    const float* Wv = (const float*)d_in[5];
    const float* bv = (const float*)d_in[6];
    const float* Wo = (const float*)d_in[7];
    const float* bo = (const float*)d_in[8];
    float* out = (float*)d_out;

    char* ws = (char*)d_ws;
    const size_t MB = 1024 * 1024;
    _Float16* xh   = (_Float16*)(ws + 0);        // 2M halves
    _Float16* wqh  = (_Float16*)(ws + 4  * MB);  // 1M halves
    _Float16* wkh  = (_Float16*)(ws + 6  * MB);
    _Float16* wvh  = (_Float16*)(ws + 8  * MB);
    _Float16* woh  = (_Float16*)(ws + 10 * MB);
    _Float16* qh   = (_Float16*)(ws + 12 * MB);  // [16][2048][64]
    _Float16* kh   = (_Float16*)(ws + 16 * MB);  // [16][2048][64]
    _Float16* vth  = (_Float16*)(ws + 20 * MB);  // [16][64][2048]
    _Float16* ctxh = (_Float16*)(ws + 24 * MB);  // [2048][1024]

    const int nX = S_LEN * HIDN;   // batch 0 only
    const int nW = HIDN * HIDN;
    cvt_f32_to_f16<<<(nX + 255) / 256, 256, 0, stream>>>(x,  xh,  nX);
    cvt_f32_to_f16<<<(nW + 255) / 256, 256, 0, stream>>>(Wq, wqh, nW);
    cvt_f32_to_f16<<<(nW + 255) / 256, 256, 0, stream>>>(Wk, wkh, nW);
    cvt_f32_to_f16<<<(nW + 255) / 256, 256, 0, stream>>>(Wv, wvh, nW);
    cvt_f32_to_f16<<<(nW + 255) / 256, 256, 0, stream>>>(Wo, woh, nW);

    dim3 ggrid(HIDN / 128, S_LEN / 64);   // (8, 32)
    // Q: fold 1/sqrt(64) into the projection (applies to bias too, matching ref)
    gemm_wmma<<<ggrid, 256, 0, stream>>>(xh, wqh, bq, qh,  S_LEN, HIDN, HIDN, 1, 0.125f);
    gemm_wmma<<<ggrid, 256, 0, stream>>>(xh, wkh, bk, kh,  S_LEN, HIDN, HIDN, 1, 1.0f);
    gemm_wmma<<<ggrid, 256, 0, stream>>>(xh, wvh, bv, vth, S_LEN, HIDN, HIDN, 2, 1.0f);

    attn_flash<<<dim3(S_LEN / 64, NHEAD), 128, 0, stream>>>(qh, kh, vth, ctxh);

    gemm_wmma<<<ggrid, 256, 0, stream>>>(ctxh, woh, bo, out, S_LEN, HIDN, HIDN, 0, 1.0f);
}
